// SharedNATransformer_48292612276771
// MI455X (gfx1250) — compile-verified, matrix operand order checked by reading, not run on hardware
//
#include <hip/hip_runtime.h>
#include <stdint.h>

// ---------------------------------------------------------------------------
// Shapes (fixed by the reference setup)
// ---------------------------------------------------------------------------
constexpr int NTOK = 1024;     // B*T*H*W = 1*4*16*16
constexpr int DM   = 768;
constexpr int NH   = 12;
constexpr int HD   = 64;
constexpr int FF   = 3072;
constexpr int QKV  = 3 * DM;   // 2304
constexpr int TT = 4, HH = 16, WW = 16, KS = 5;

#define DEV __device__ __forceinline__

typedef __attribute__((ext_vector_type(16))) __bf16 bf16x16;
typedef __attribute__((ext_vector_type(8)))  float  f32x8;

union FragBF { uint32_t u[8]; bf16x16 v; };

DEV float bf2f(uint16_t h) {
    uint32_t u = ((uint32_t)h) << 16;
    float f; __builtin_memcpy(&f, &u, 4); return f;
}
DEV uint16_t f2bf(float f) {
    uint32_t u; __builtin_memcpy(&u, &f, 4);
    u += 0x7FFFu + ((u >> 16) & 1u);           // round-to-nearest-even
    return (uint16_t)(u >> 16);
}
DEV int imin(int a, int b) { return a < b ? a : b; }
DEV int imax(int a, int b) { return a > b ? a : b; }

// ---------------------------------------------------------------------------
// CDNA5 async global->LDS copy (ASYNCcnt-tracked), with sync fallback.
// Builtin signature (from hipcc diagnostic): v4i addrspace(1)*, v4i addrspace(3)*,
// imm offset, imm cpol.
// ---------------------------------------------------------------------------
#if defined(__has_builtin)
#  if __has_builtin(__builtin_amdgcn_global_load_async_to_lds_b128)
#    define ASYNC_LDS 1
#  endif
#endif

#ifdef ASYNC_LDS
typedef int v4i __attribute__((vector_size(16)));
typedef __attribute__((address_space(1))) v4i gbl_v4i;
typedef __attribute__((address_space(3))) v4i lds_v4i;
DEV void async_cp_b128(const void* g, void* l) {
    __builtin_amdgcn_global_load_async_to_lds_b128(
        (gbl_v4i*)g, (lds_v4i*)l, 0, 0);
}
#endif

// ---------------------------------------------------------------------------
// fp32 -> bf16 conversion (weights, once per call; trivially cheap vs GEMMs)
// ---------------------------------------------------------------------------
__global__ void cvt_f32_bf16(const float* __restrict__ s, uint16_t* __restrict__ d, int n) {
    int i = blockIdx.x * blockDim.x + threadIdx.x;
    int stride = gridDim.x * blockDim.x;
    for (; i < n; i += stride) d[i] = f2bf(s[i]);
}

// ---------------------------------------------------------------------------
// RMSNorm: one 256-thread block per token (768 = 3 elems/thread).
// Emits bf16 (for GEMM input) or fp32 (final output).
// ---------------------------------------------------------------------------
__global__ __launch_bounds__(256)
void rmsnorm_kernel(const float* __restrict__ x, const float* __restrict__ w,
                    uint16_t* __restrict__ outB, float* __restrict__ outF) {
    __shared__ float red[8];
    const int tok = blockIdx.x;
    const float* xr = x + (size_t)tok * DM;
    const int t = threadIdx.x;
    float v0 = xr[t], v1 = xr[t + 256], v2 = xr[t + 512];
    float ss = v0 * v0 + v1 * v1 + v2 * v2;
    #pragma unroll
    for (int off = 16; off > 0; off >>= 1) ss += __shfl_xor(ss, off, 32);
    if ((t & 31) == 0) red[t >> 5] = ss;
    __syncthreads();
    float tot = 0.f;
    #pragma unroll
    for (int i = 0; i < 8; i++) tot += red[i];
    const float inv = rsqrtf(tot / (float)DM + 1e-6f);
    if (outB) {
        outB[(size_t)tok * DM + t      ] = f2bf(v0 * inv * w[t]);
        outB[(size_t)tok * DM + t + 256] = f2bf(v1 * inv * w[t + 256]);
        outB[(size_t)tok * DM + t + 512] = f2bf(v2 * inv * w[t + 512]);
    } else {
        outF[(size_t)tok * DM + t      ] = v0 * inv * w[t];
        outF[(size_t)tok * DM + t + 256] = v1 * inv * w[t + 256];
        outF[(size_t)tok * DM + t + 512] = v2 * inv * w[t + 512];
    }
}

// ---------------------------------------------------------------------------
// bf16 WMMA GEMM:  C[n][m] = sum_k A[n][k] * W[m][k]  (+bias[m]) (+res, fp32 out)
// Block tile 128x128, 8 waves (wave32) as 4x2, each wave a 32x64 tile:
// 8 v_wmma_f32_16x16x32_bf16 per K-step from 2 A-frags + 4 B-frags.
// Double-buffered LDS (one barrier per K-step); next tile copied with
// GLOBAL_LOAD_ASYNC_TO_LDS_B128 (ASYNCcnt) when available, overlapping WMMA.
// ---------------------------------------------------------------------------
DEV f32x8 wmma_bf16(const FragBF& a, const FragBF& b, f32x8 c) {
    return __builtin_amdgcn_wmma_f32_16x16x32_bf16(
        /*neg_a=*/false, a.v, /*neg_b=*/false, b.v,
        /*c_mod=*/(short)0, c, /*reuse_a=*/false, /*reuse_b=*/false);
}

__global__ __launch_bounds__(256)
void gemm_bf16_wmma(const uint16_t* __restrict__ A, const uint16_t* __restrict__ W,
                    const float* __restrict__ bias, const float* __restrict__ res,
                    float* __restrict__ outF, uint16_t* __restrict__ outB,
                    int K, int M) {
    __shared__ __align__(16) uint16_t As[2][128][40];   // padded rows (80B)
    __shared__ __align__(16) uint16_t Bs[2][128][40];

    const int lane = threadIdx.x & 31;
    const int wave = threadIdx.x >> 5;
    const int wm = wave & 3;           // 4 waves down  (32 rows each)
    const int wn = wave >> 2;          // 2 waves across (64 cols each)
    const int rowBase = blockIdx.y * 128;
    const int colBase = blockIdx.x * 128;

    // per-thread global->LDS assignment: 16 bf16 (2 x b128) per matrix
    const int trow = threadIdx.x >> 1;             // 0..127
    const int tkc  = (threadIdx.x & 1) * 16;       // 0 or 16 (bf16 elems)
    const uint16_t* Ag = A + (size_t)(rowBase + trow) * K + tkc;
    const uint16_t* Bg = W + (size_t)(colBase + trow) * K + tkc;

    // WMMA A/B fragment K mapping (ISA 7.12.2, 16-bit 16x32):
    // VGPR v holds K pair (v<4 ? 2v : 16+2(v-4)) (+8 for lanes 16..31)
    int kks[8];
    #pragma unroll
    for (int v = 0; v < 8; v++)
        kks[v] = ((v < 4) ? 2 * v : 16 + 2 * (v - 4)) + ((lane & 16) ? 8 : 0);
    const int mA0 = wm * 32 + (lane & 15);
    const int nB0 = wn * 64 + (lane & 15);

    f32x8 acc[8] = { {}, {}, {}, {}, {}, {}, {}, {} };   // [hm*4+hn]

#ifndef ASYNC_LDS
    uint4 ra0, ra1, rb0, rb1;                            // staging regs
#endif

    auto issue_fetch = [&](int k0, int buf) {
#ifdef ASYNC_LDS
        async_cp_b128(Ag + k0,     &As[buf][trow][tkc]);
        async_cp_b128(Ag + k0 + 8, &As[buf][trow][tkc + 8]);
        async_cp_b128(Bg + k0,     &Bs[buf][trow][tkc]);
        async_cp_b128(Bg + k0 + 8, &Bs[buf][trow][tkc + 8]);
#else
        const uint4* ap = (const uint4*)(Ag + k0);
        const uint4* bp = (const uint4*)(Bg + k0);
        ra0 = ap[0]; ra1 = ap[1];
        rb0 = bp[0]; rb1 = bp[1];
#endif
        if (k0 + 32 < K) {
            __builtin_prefetch(Ag + k0 + 32, 0, 1);
            __builtin_prefetch(Bg + k0 + 32, 0, 1);
        }
    };
#ifndef ASYNC_LDS
    auto commit_fetch = [&](int buf) {
        *(uint4*)&As[buf][trow][tkc]     = ra0;
        *(uint4*)&As[buf][trow][tkc + 8] = ra1;
        *(uint4*)&Bs[buf][trow][tkc]     = rb0;
        *(uint4*)&Bs[buf][trow][tkc + 8] = rb1;
    };
#endif
    auto fetch_done_barrier = [&]() {
#ifdef ASYNC_LDS
        asm volatile("s_wait_asynccnt 0" ::: "memory");
#endif
        __syncthreads();
    };

    // prologue: tile 0 -> buffer 0
    issue_fetch(0, 0);
#ifndef ASYNC_LDS
    commit_fetch(0);
#endif
    fetch_done_barrier();

    int cur = 0;
    for (int k0 = 32; k0 <= K; k0 += 32) {
        const int nxt = cur ^ 1;
        const bool more = (k0 < K);
        if (more) issue_fetch(k0, nxt);

        FragBF a0, a1, b[4];
        #pragma unroll
        for (int v = 0; v < 8; v++) {
            a0.u[v]   = *(const uint32_t*)&As[cur][mA0     ][kks[v]];
            a1.u[v]   = *(const uint32_t*)&As[cur][mA0 + 16][kks[v]];
            b[0].u[v] = *(const uint32_t*)&Bs[cur][nB0     ][kks[v]];
            b[1].u[v] = *(const uint32_t*)&Bs[cur][nB0 + 16][kks[v]];
            b[2].u[v] = *(const uint32_t*)&Bs[cur][nB0 + 32][kks[v]];
            b[3].u[v] = *(const uint32_t*)&Bs[cur][nB0 + 48][kks[v]];
        }
        #pragma unroll
        for (int j = 0; j < 4; j++) {
            acc[j]     = wmma_bf16(a0, b[j], acc[j]);
            acc[4 + j] = wmma_bf16(a1, b[j], acc[4 + j]);
        }

#ifndef ASYNC_LDS
        if (more) commit_fetch(nxt);
#endif
        fetch_done_barrier();
        cur = nxt;
    }

    // D layout: VGPR r -> M = r (+8 for lanes 16..31); N = lane&15
    #pragma unroll
    for (int tidx = 0; tidx < 8; tidx++) {
        const int hm = tidx >> 2, hn = tidx & 3;
        const int row0 = rowBase + wm * 32 + hm * 16 + ((lane & 16) ? 8 : 0);
        const int col  = colBase + wn * 64 + hn * 16 + (lane & 15);
        const float bv = bias ? bias[col] : 0.f;
        #pragma unroll
        for (int r = 0; r < 8; r++) {
            const int row = row0 + r;
            float val = acc[tidx][r] + bv;
            const size_t idx = (size_t)row * M + col;
            if (outF) outF[idx] = res[idx] + val;
            else      outB[idx] = f2bf(val);
        }
    }
}

// ---------------------------------------------------------------------------
// NA3D attention: one wave32 per (token, head). Lane handles 2 head dims.
// 125 neighbors, online softmax, wave shuffle dot-product reduction.
// ---------------------------------------------------------------------------
__global__ __launch_bounds__(256)
void na3d_attn(const uint16_t* __restrict__ qkv, uint16_t* __restrict__ attn_out) {
    const int g = blockIdx.x * 8 + (threadIdx.x >> 5);   // wave id in grid
    const int n = g / NH;
    const int e = g - n * NH;
    const int lane = threadIdx.x & 31;
    const int d0 = lane * 2;

    const int t  = n >> 8;
    const int hh = (n >> 4) & 15;
    const int ww = n & 15;

    const size_t qb = (size_t)n * QKV + e * HD;
    float q0 = bf2f(qkv[qb + d0])     * 0.125f;   // 1/sqrt(64)
    float q1 = bf2f(qkv[qb + d0 + 1]) * 0.125f;

    float m = -3.0e38f, l = 0.f, a0 = 0.f, a1 = 0.f;
    const int hstart = imin(imax(hh - 2, 0), HH - KS);
    const int wstart = imin(imax(ww - 2, 0), WW - KS);

    for (int kt = 0; kt < KS; kt++) {
        const int rt = t - (KS - 1) + kt;         // causal in T
        const bool valid = (rt >= 0);
        const int it = imin(imax(rt, 0), TT - 1);
        for (int kh = 0; kh < KS; kh++) {
            const int ih = hstart + kh;
            for (int kw = 0; kw < KS; kw++) {
                const int iw = wstart + kw;
                const int nn = (it << 8) + (ih << 4) + iw;
                const size_t kb = (size_t)nn * QKV + DM + e * HD;
                float s = q0 * bf2f(qkv[kb + d0]) + q1 * bf2f(qkv[kb + d0 + 1]);
                #pragma unroll
                for (int off = 16; off > 0; off >>= 1) s += __shfl_xor(s, off, 32);
                if (!valid) s = -1.0e9f;
                const float mn   = fmaxf(m, s);
                const float corr = __expf(m - mn);
                const float p    = __expf(s - mn);
                const size_t vb = (size_t)nn * QKV + 2 * DM + e * HD;
                a0 = a0 * corr + p * bf2f(qkv[vb + d0]);
                a1 = a1 * corr + p * bf2f(qkv[vb + d0 + 1]);
                l = l * corr + p;
                m = mn;
            }
        }
    }
    const float invl = 1.f / l;
    attn_out[(size_t)n * DM + e * HD + d0]     = f2bf(a0 * invl);
    attn_out[(size_t)n * DM + e * HD + d0 + 1] = f2bf(a1 * invl);
}

// ---------------------------------------------------------------------------
// SwiGLU gate: g = silu(h1) * h3  (elementwise, bf16)
// ---------------------------------------------------------------------------
__global__ void silu_mul(const uint16_t* __restrict__ h1, const uint16_t* __restrict__ h3,
                         uint16_t* __restrict__ g, int n) {
    int i = blockIdx.x * blockDim.x + threadIdx.x;
    int stride = gridDim.x * blockDim.x;
    for (; i < n; i += stride) {
        float a = bf2f(h1[i]);
        float b = bf2f(h3[i]);
        float s = a / (1.f + __expf(-a));
        g[i] = f2bf(s * b);
    }
}

// ---------------------------------------------------------------------------
// Host orchestration
// ---------------------------------------------------------------------------
extern "C" void kernel_launch(void* const* d_in, const int* in_sizes, int n_in,
                              void* d_out, int out_size, void* d_ws, size_t ws_size,
                              hipStream_t stream) {
    const float* x_in  = (const float*)d_in[0];
    const float* norm1 = (const float*)d_in[4];
    const float* norm2 = (const float*)d_in[5];
    const float* qkv_w = (const float*)d_in[6];
    const float* qkv_b = (const float*)d_in[7];
    const float* out_w = (const float*)d_in[8];
    const float* out_b = (const float*)d_in[9];
    const float* w1    = (const float*)d_in[10];
    const float* w3    = (const float*)d_in[11];
    const float* w2    = (const float*)d_in[12];
    const float* fnorm = (const float*)d_in[13];

    char* p = (char*)d_ws;
    auto alloc = [&](size_t bytes) -> void* {
        void* r = (void*)p;
        p += (bytes + 255) & ~(size_t)255;
        return r;
    };

    uint16_t* wq   = (uint16_t*)alloc((size_t)2 * QKV * DM * 2);
    uint16_t* wo   = (uint16_t*)alloc((size_t)2 * DM * DM * 2);
    uint16_t* w1b  = (uint16_t*)alloc((size_t)2 * FF * DM * 2);
    uint16_t* w3b  = (uint16_t*)alloc((size_t)2 * FF * DM * 2);
    uint16_t* w2b  = (uint16_t*)alloc((size_t)2 * DM * FF * 2);
    float*    xb   = (float*)   alloc((size_t)NTOK * DM * 4);
    uint16_t* xn   = (uint16_t*)alloc((size_t)NTOK * DM * 2);
    uint16_t* qkvb = (uint16_t*)alloc((size_t)NTOK * QKV * 2);
    uint16_t* att  = (uint16_t*)alloc((size_t)NTOK * DM * 2);
    uint16_t* h1   = (uint16_t*)alloc((size_t)NTOK * FF * 2);
    uint16_t* h3   = (uint16_t*)alloc((size_t)NTOK * FF * 2);
    uint16_t* g    = (uint16_t*)alloc((size_t)NTOK * FF * 2);

    cvt_f32_bf16<<<1024, 256, 0, stream>>>(qkv_w, wq,  2 * QKV * DM);
    cvt_f32_bf16<<<1024, 256, 0, stream>>>(out_w, wo,  2 * DM * DM);
    cvt_f32_bf16<<<1024, 256, 0, stream>>>(w1,    w1b, 2 * FF * DM);
    cvt_f32_bf16<<<1024, 256, 0, stream>>>(w3,    w3b, 2 * FF * DM);
    cvt_f32_bf16<<<1024, 256, 0, stream>>>(w2,    w2b, 2 * DM * FF);

    (void)hipMemcpyAsync(xb, x_in, (size_t)NTOK * DM * 4, hipMemcpyDeviceToDevice, stream);

    for (int l = 0; l < 2; l++) {
        // --- attention block ---
        rmsnorm_kernel<<<NTOK, 256, 0, stream>>>(xb, norm1 + l * DM, xn, nullptr);
        gemm_bf16_wmma<<<dim3(QKV / 128, NTOK / 128), 256, 0, stream>>>(
            xn, wq + (size_t)l * QKV * DM, qkv_b + l * QKV,
            nullptr, nullptr, qkvb, DM, QKV);
        na3d_attn<<<NTOK * NH / 8, 256, 0, stream>>>(qkvb, att);
        gemm_bf16_wmma<<<dim3(DM / 128, NTOK / 128), 256, 0, stream>>>(
            att, wo + (size_t)l * DM * DM, out_b + l * DM,
            xb, xb, nullptr, DM, DM);                         // fused residual, fp32 out
        // --- FFN block ---
        rmsnorm_kernel<<<NTOK, 256, 0, stream>>>(xb, norm2 + l * DM, xn, nullptr);
        gemm_bf16_wmma<<<dim3(FF / 128, NTOK / 128), 256, 0, stream>>>(
            xn, w1b + (size_t)l * FF * DM, nullptr, nullptr, nullptr, h1, DM, FF);
        gemm_bf16_wmma<<<dim3(FF / 128, NTOK / 128), 256, 0, stream>>>(
            xn, w3b + (size_t)l * FF * DM, nullptr, nullptr, nullptr, h3, DM, FF);
        silu_mul<<<1024, 256, 0, stream>>>(h1, h3, g, NTOK * FF);
        gemm_bf16_wmma<<<dim3(DM / 128, NTOK / 128), 256, 0, stream>>>(
            g, w2b + (size_t)l * DM * FF, nullptr, xb, xb, nullptr, FF, DM);
    }
    rmsnorm_kernel<<<NTOK, 256, 0, stream>>>(xb, fnorm, nullptr, (float*)d_out);
}
